// Q_SelfAttention_90967407329788
// MI455X (gfx1250) — compile-verified, matrix operand order checked by reading, not run on hardware
//
#include <hip/hip_runtime.h>
#include <hip/hip_bf16.h>
#include <math.h>

// ---------------------------------------------------------------------------
// CDNA5 (gfx1250) wave32 WMMA implementation of quadrangle window attention.
// All GEMM/attention operands flow as bf16 (qkv intermediate = 115 MB, fits
// in the 192 MB L2, making the bilinear K/V gather L2-resident).
// ---------------------------------------------------------------------------

typedef __attribute__((ext_vector_type(16))) __bf16 v16bf;
typedef __attribute__((ext_vector_type(4)))  __bf16 bf16x4;
typedef __attribute__((ext_vector_type(8)))  float  v8f;

#define NHEAD 12
#define WS    7
#define CDIM  768
#define HW    56
#define NPIX  3136          // 56*56
#define BATCH 8
#define DHEAD 64
#define WN    8             // windows per side
#define C3    2304          // 3*C

// pack two floats to packed bf16 (RNE) -> one 32-bit word
__device__ __forceinline__ unsigned pk2(float a, float b) {
  union { __bf16 h[2]; unsigned u; } t;
  t.h[0] = (__bf16)a;
  t.h[1] = (__bf16)b;
  return t.u;
}

// ---- WMMA fragment loaders (per ISA 7.12.2, wave32) -----------------------
// A (MxK, 16x32 bf16) / B-as-NxK: lane l<16 holds row l, K in {0..7,16..23};
// lane l+16 holds row l, K in {8..15,24..31}.  16 elements per lane.
__device__ __forceinline__ v16bf frag_rowmajor(const __bf16* base, int stride) {
  const int lane = threadIdx.x & 31;
  const int r = lane & 15;
  const int h = lane >> 4;
  const __bf16* p = base + r * stride + 8 * h;
  v16bf f;
#pragma unroll
  for (int j = 0; j < 8; ++j) f[j] = p[j];
#pragma unroll
  for (int j = 0; j < 8; ++j) f[j + 8] = p[16 + j];
  return f;
}

// B fragment when source is stored K-major (KxN row-major): B[k][n].
__device__ __forceinline__ v16bf frag_kxn(const __bf16* base, int stride) {
  const int lane = threadIdx.x & 31;
  const int n = lane & 15;
  const int h = lane >> 4;
  const __bf16* p = base + 8 * h * stride + n;
  v16bf f;
#pragma unroll
  for (int j = 0; j < 8; ++j) f[j] = p[j * stride];
#pragma unroll
  for (int j = 0; j < 8; ++j) f[j + 8] = p[(16 + j) * stride];
  return f;
}

// ---------------------------------------------------------------------------
// Kernel 0: streaming f32 -> bf16 conversion (vectorized 4-wide)
// ---------------------------------------------------------------------------
__global__ __launch_bounds__(256) void f32_to_bf16(
    const float* __restrict__ in, __bf16* __restrict__ out, int n4) {
  int i = blockIdx.x * 256 + threadIdx.x;
  const int stride = gridDim.x * 256;
  for (; i < n4; i += stride) {
    float4 v = ((const float4*)in)[i];
    uint2 p; p.x = pk2(v.x, v.y); p.y = pk2(v.z, v.w);
    ((uint2*)out)[i] = p;
  }
}

// ---------------------------------------------------------------------------
// Kernel 1: Out[m][n] = sum_k A[m][k]*W[n][k] + bias[n]   (bf16 WMMA, f32 acc)
// Block = 256 threads = 8 waves; block tile 128(M) x 32(N); K step 32.
// A/W already bf16 -> staging is pure b128 global load -> b128 LDS store.
// ---------------------------------------------------------------------------
template <typename OutT>
__global__ __launch_bounds__(256) void gemm_bf16_bias(
    const __bf16* __restrict__ A, const __bf16* __restrict__ W,
    const float* __restrict__ bias, OutT* __restrict__ Out,
    int M, int N, int K) {
  __shared__ __bf16 As[128][40];   // padded stride vs bank conflicts
  __shared__ __bf16 Wsh[32][40];
  const int tid  = threadIdx.x;
  const int wave = tid >> 5;
  const int lane = tid & 31;
  const int n0 = blockIdx.x * 32;
  const int m0 = blockIdx.y * 128;

  // hoisted per-thread staging pointers; uint4 = 8 bf16; K step = 4 uint4
  const uint4* aSrc[2];
  uint4* aDst[2];
#pragma unroll
  for (int u = 0; u < 2; ++u) {
    int f = tid + u * 256;            // uint4 index into 128 x 4 tile
    int r = f >> 2, c8 = (f & 3) * 8;
    aSrc[u] = (const uint4*)(A + (size_t)(m0 + r) * K + c8);
    aDst[u] = (uint4*)&As[r][c8];
  }
  const int wr = tid >> 2, wc8 = (tid & 3) * 8;   // 32 rows x 4 uint4 (tid<128)
  const uint4* wSrc = (const uint4*)(W + (size_t)(n0 + wr) * K + wc8);
  uint4* wDst = (uint4*)&Wsh[wr][wc8];

  v8f acc0 = {}, acc1 = {};
  for (int kk = 0; kk < K; kk += 32) {
#pragma unroll
    for (int u = 0; u < 2; ++u) {
      uint4 t = *aSrc[u];
      __builtin_prefetch(aSrc[u] + 4, 0, 1);      // next K tile
      *aDst[u] = t;
      aSrc[u] += 4;
    }
    if (tid < 128) {
      uint4 t = *wSrc;
      *wDst = t;
      wSrc += 4;
    }
    __syncthreads();
    v16bf a  = frag_rowmajor(&As[wave * 16][0], 40);
    v16bf b0 = frag_rowmajor(&Wsh[0][0], 40);
    v16bf b1 = frag_rowmajor(&Wsh[16][0], 40);
    acc0 = __builtin_amdgcn_wmma_f32_16x16x32_bf16(false, a, false, b0,
                                                   (short)0, acc0, false, false);
    acc1 = __builtin_amdgcn_wmma_f32_16x16x32_bf16(false, a, false, b1,
                                                   (short)0, acc1, false, false);
    __syncthreads();
  }
  const int h  = lane >> 4;
  const int ln = lane & 15;
  const float bv0 = bias[n0 + ln];
  const float bv1 = bias[n0 + 16 + ln];
#pragma unroll
  for (int r = 0; r < 8; ++r) {
    int row = m0 + wave * 16 + r + 8 * h;
    Out[(size_t)row * N + n0 + ln]      = (OutT)(acc0[r] + bv0);
    Out[(size_t)row * N + n0 + 16 + ln] = (OutT)(acc1[r] + bv1);
  }
}

// ---------------------------------------------------------------------------
// Kernel 2: per (b, window): avg-pool -> leaky relu -> 108-ch conv ->
// per-head projective affine -> 49 sampling coords. Tiny VALU kernel.
// ---------------------------------------------------------------------------
__global__ __launch_bounds__(256) void coords_kernel(
    const float* __restrict__ x, const float* __restrict__ conv_w,
    const float* __restrict__ conv_b, float* __restrict__ coords) {
  __shared__ float act[CDIM];
  __shared__ float samp[NHEAD * 9];
  __shared__ float aff[NHEAD][9];
  const int tid = threadIdx.x;
  const int win = blockIdx.x;          // 0..63
  const int b   = blockIdx.y;          // 0..7
  const int wh  = win >> 3, ww = win & 7;

  // 7x7 average pool + leaky relu (0.01)
  for (int c = tid; c < CDIM; c += 256) {
    float s = 0.f;
    for (int i = 0; i < WS; ++i) {
      const float* row =
          x + ((size_t)b * NPIX + (size_t)(wh * WS + i) * HW + ww * WS) * CDIM + c;
      for (int j = 0; j < WS; ++j) s += row[(size_t)j * CDIM];
    }
    s *= (1.f / 49.f);
    act[c] = (s >= 0.f) ? s : 0.01f * s;
  }
  __syncthreads();

  // samp = act . conv_w^T + conv_b   (108 outputs)
  for (int o = tid; o < NHEAD * 9; o += 256) {
    float s = conv_b[o];
    const float* wrow = conv_w + (size_t)o * CDIM;
    for (int c = 0; c < CDIM; ++c) s += act[c] * wrow[c];
    samp[o] = s;
  }
  __syncthreads();

  // per-head 3x3 affine = [[rot@shear@scale, off],[pj0,pj1,1]]
  if (tid < NHEAD) {
    const float* s = &samp[tid * 9];
    float offx = s[0] / (float)WN, offy = s[1] / (float)WN;
    float sc0 = s[2] + 1.f, sc1 = s[3] + 1.f;
    float sh0 = s[4], sh1 = s[5];
    float cr = cosf(s[8]), sr = sinf(s[8]);
    aff[tid][0] = sc0 * (cr + sr * sh1);
    aff[tid][1] = sc1 * (cr * sh0 + sr);
    aff[tid][2] = offx;
    aff[tid][3] = sc0 * (cr * sh1 - sr);
    aff[tid][4] = sc1 * (cr - sr * sh0);
    aff[tid][5] = offy;
    aff[tid][6] = s[6];
    aff[tid][7] = s[7];
    aff[tid][8] = 1.f;
  }
  __syncthreads();

  const float wcenx = -1.f + 2.f * (float)(ww * WS + 3) / (float)(HW - 1);
  const float wceny = -1.f + 2.f * (float)(wh * WS + 3) / (float)(HW - 1);
  for (int t = tid; t < NHEAD * WS * WS; t += 256) {
    int g = t / 49, n = t % 49;
    int wi = n / 7, wj = n % 7;
    float wx = 2.f * (float)(wj - 3) / (float)(HW - 1);
    float wy = 2.f * (float)(wi - 3) / (float)(HW - 1);
    const float* Am = aff[g];
    float tx = Am[0] * wx + Am[1] * wy + Am[2];
    float ty = Am[3] * wx + Am[4] * wy + Am[5];
    float t3 = Am[6] * wx + Am[7] * wy + Am[8];
    if (t3 == 0.f) t3 = 1e-6f;
    float cx = wcenx + tx / t3;
    float cy = wceny + ty / t3;
    size_t base =
        (((size_t)(b * NHEAD + g) * HW + (wh * WS + wi)) * HW + (ww * WS + wj)) * 2;
    coords[base]     = cx;
    coords[base + 1] = cy;
  }
}

// ---------------------------------------------------------------------------
// Kernel 3: fused bilinear grid-sample of K/V + window attention (bf16 qkv).
// One block per (head, window, batch). 49 tokens padded to 64 => 4x4 tiles of
// 16x16 WMMA, K=64 (two 16x16x32 bf16 WMMA steps).
// ---------------------------------------------------------------------------
__global__ __launch_bounds__(256) void attn_kernel(
    const __bf16* __restrict__ qkv, const float* __restrict__ coords,
    __bf16* __restrict__ outp) {
  __shared__ __bf16 Qs[64][72];     // Q tile, later reused to hold attn probs
  __shared__ __bf16 Ks[64][72];
  __shared__ __bf16 Vs[64][72];
  __shared__ float  S[64][65];      // f32 scores
  __shared__ int    cpix[4][49];
  __shared__ float  cwgt[4][49];
  __shared__ float  rmax[64], rinv[64];

  const int tid = threadIdx.x;
  const int g   = blockIdx.x;       // head
  const int win = blockIdx.y;       // window in image
  const int b   = blockIdx.z;
  const int wh  = win >> 3, ww = win & 7;
  const int bg  = b * NHEAD + g;

  // 1. per-position bilinear corner indices/weights (zero-pad outside image)
  if (tid < 49) {
    int wi = tid / 7, wj = tid % 7;
    size_t cidx =
        (((size_t)bg * HW + (wh * WS + wi)) * HW + (ww * WS + wj)) * 2;
    float gx = coords[cidx], gy = coords[cidx + 1];
    float ix = (gx + 1.f) * 0.5f * (float)(HW - 1);
    float iy = (gy + 1.f) * 0.5f * (float)(HW - 1);
    float x0 = floorf(ix), y0 = floorf(iy);
    float wx1 = ix - x0, wy1 = iy - y0;
    float xs[2]  = {x0, x0 + 1.f};
    float ys[2]  = {y0, y0 + 1.f};
    float wxs[2] = {1.f - wx1, wx1};
    float wys[2] = {1.f - wy1, wy1};
#pragma unroll
    for (int cy = 0; cy < 2; ++cy)
#pragma unroll
      for (int cx = 0; cx < 2; ++cx) {
        float xc = xs[cx], yc = ys[cy];
        float valid = (xc >= 0.f && xc <= (float)(HW - 1) &&
                       yc >= 0.f && yc <= (float)(HW - 1)) ? 1.f : 0.f;
        int xi = (int)fminf(fmaxf(xc, 0.f), (float)(HW - 1));
        int yi = (int)fminf(fmaxf(yc, 0.f), (float)(HW - 1));
        cpix[cy * 2 + cx][tid] = yi * HW + xi;
        cwgt[cy * 2 + cx][tid] = wxs[cx] * wys[cy] * valid;
      }
  }
  __syncthreads();

  // 2. stage Q (raw bf16 copy) and K/V (bilinear gather, f32 accumulate),
  //    vectorized 4 channels/thread; gather is L2-resident (qkv bf16 < L2)
  const size_t qkvbase = (size_t)b * NPIX * C3;
  const int ch_q = g * DHEAD;
  for (int i = tid; i < 64 * 16; i += 256) {
    int n = i >> 4, d4 = (i & 15) * 4;
    uint2 pq = {0u, 0u};
    float4 kg = {0.f, 0.f, 0.f, 0.f};
    float4 vg = {0.f, 0.f, 0.f, 0.f};
    if (n < 49) {
      int wi = n / 7, wj = n % 7;
      int pix = (wh * WS + wi) * HW + (ww * WS + wj);
      pq = *(const uint2*)(qkv + qkvbase + (size_t)pix * C3 + ch_q + d4);
#pragma unroll
      for (int c = 0; c < 4; ++c) {
        float wgt = cwgt[c][n];
        const __bf16* p = qkv + qkvbase + (size_t)cpix[c][n] * C3 + ch_q + d4;
        bf16x4 kc = *(const bf16x4*)(p + CDIM);        // K channels at +768
        bf16x4 vc = *(const bf16x4*)(p + 2 * CDIM);    // V channels at +1536
        kg.x += wgt * (float)kc[0]; kg.y += wgt * (float)kc[1];
        kg.z += wgt * (float)kc[2]; kg.w += wgt * (float)kc[3];
        vg.x += wgt * (float)vc[0]; vg.y += wgt * (float)vc[1];
        vg.z += wgt * (float)vc[2]; vg.w += wgt * (float)vc[3];
      }
    }
    uint2 pkk; pkk.x = pk2(kg.x, kg.y); pkk.y = pk2(kg.z, kg.w);
    uint2 pv;  pv.x  = pk2(vg.x, vg.y); pv.y  = pk2(vg.z, vg.w);
    *(uint2*)&Qs[n][d4] = pq;
    *(uint2*)&Ks[n][d4] = pkk;
    *(uint2*)&Vs[n][d4] = pv;
  }
  __syncthreads();

  // 3. S = (Q @ K^T) * D^-0.5 : 4x4 tiles, 2 tiles per wave
  const int wave = tid >> 5, lane = tid & 31;
  const int lh = lane >> 4, ln = lane & 15;
#pragma unroll
  for (int t = 0; t < 2; ++t) {
    int tile = wave * 2 + t;
    int tm = tile >> 2, tn = tile & 3;
    v8f acc = {};
#pragma unroll
    for (int kk = 0; kk < 64; kk += 32) {
      v16bf a  = frag_rowmajor(&Qs[tm * 16][kk], 72);
      v16bf bb = frag_rowmajor(&Ks[tn * 16][kk], 72);
      acc = __builtin_amdgcn_wmma_f32_16x16x32_bf16(false, a, false, bb,
                                                    (short)0, acc, false, false);
    }
#pragma unroll
    for (int r = 0; r < 8; ++r)
      S[tm * 16 + r + 8 * lh][tn * 16 + ln] = acc[r] * 0.125f;  // 64^-0.5
  }
  __syncthreads();

  // 4. masked softmax over the 49 valid keys; probs -> packed bf16 in Qs
  if (tid < 49) {
    float m = -1e30f;
    for (int c = 0; c < 49; ++c) m = fmaxf(m, S[tid][c]);
    float s = 0.f;
    for (int c = 0; c < 49; ++c) s += __expf(S[tid][c] - m);
    rmax[tid] = m;
    rinv[tid] = 1.f / s;
  }
  __syncthreads();
  for (int i = tid; i < 64 * 32; i += 256) {
    int r = i >> 5, c2 = (i & 31) * 2;
    float a0 = 0.f, a1 = 0.f;
    if (r < 49) {
      if (c2 < 49)     a0 = __expf(S[r][c2]     - rmax[r]) * rinv[r];
      if (c2 + 1 < 49) a1 = __expf(S[r][c2 + 1] - rmax[r]) * rinv[r];
    }
    *(unsigned*)&Qs[r][c2] = pk2(a0, a1);
  }
  __syncthreads();

  // 5. O = attn @ V, scatter valid rows back to bf16 [B, N, C]
#pragma unroll
  for (int t = 0; t < 2; ++t) {
    int tile = wave * 2 + t;
    int tm = tile >> 2, tn = tile & 3;
    v8f acc = {};
#pragma unroll
    for (int kk = 0; kk < 64; kk += 32) {
      v16bf a  = frag_rowmajor(&Qs[tm * 16][kk], 72);
      v16bf bb = frag_kxn(&Vs[kk][tn * 16], 72);
      acc = __builtin_amdgcn_wmma_f32_16x16x32_bf16(false, a, false, bb,
                                                    (short)0, acc, false, false);
    }
#pragma unroll
    for (int r = 0; r < 8; ++r) {
      int n = tm * 16 + r + 8 * lh;
      if (n < 49) {
        int wi = n / 7, wj = n % 7;
        int pix = (wh * WS + wi) * HW + (ww * WS + wj);
        outp[((size_t)b * NPIX + pix) * CDIM + g * DHEAD + tn * 16 + ln] =
            (__bf16)acc[r];
      }
    }
  }
}

// ---------------------------------------------------------------------------
extern "C" void kernel_launch(void* const* d_in, const int* in_sizes, int n_in,
                              void* d_out, int out_size, void* d_ws, size_t ws_size,
                              hipStream_t stream) {
  const float* x      = (const float*)d_in[0];
  const float* qkv_w  = (const float*)d_in[1];
  const float* qkv_b  = (const float*)d_in[2];
  const float* proj_w = (const float*)d_in[3];
  const float* proj_b = (const float*)d_in[4];
  const float* conv_w = (const float*)d_in[5];
  const float* conv_b = (const float*)d_in[6];
  float* out = (float*)d_out;

  const int M = BATCH * NPIX;  // 25088

  // workspace layout (all offsets 256-aligned by construction)
  char* ws = (char*)d_ws;
  size_t off = 0;
  __bf16* xb = (__bf16*)(ws + off);      off += (size_t)M * CDIM * 2;          // 38.5 MB
  __bf16* qwb = (__bf16*)(ws + off);     off += (size_t)C3 * CDIM * 2;         // 3.5 MB
  __bf16* pwb = (__bf16*)(ws + off);     off += (size_t)CDIM * CDIM * 2;       // 1.2 MB
  __bf16* qkv = (__bf16*)(ws + off);     off += (size_t)M * C3 * 2;            // 115.6 MB
  float*  coords = (float*)(ws + off);   off += (size_t)BATCH * NHEAD * HW * HW * 2 * 4;
  __bf16* attn_out = (__bf16*)(ws + off);                                       // 38.5 MB

  // 0. one-time f32 -> bf16 conversions (streaming)
  {
    int n4 = M * CDIM / 4;
    int grid = (n4 + 255) / 256; if (grid > 2048) grid = 2048;
    f32_to_bf16<<<grid, 256, 0, stream>>>(x, xb, n4);
  }
  {
    int n4 = C3 * CDIM / 4;
    int grid = (n4 + 255) / 256; if (grid > 2048) grid = 2048;
    f32_to_bf16<<<grid, 256, 0, stream>>>(qkv_w, qwb, n4);
  }
  {
    int n4 = CDIM * CDIM / 4;
    int grid = (n4 + 255) / 256; if (grid > 2048) grid = 2048;
    f32_to_bf16<<<grid, 256, 0, stream>>>(proj_w, pwb, n4);
  }

  // 1. qkv = x @ qkv_w^T + qkv_b   (bf16 out)
  gemm_bf16_bias<__bf16><<<dim3(C3 / 32, M / 128), 256, 0, stream>>>(
      xb, qwb, qkv_b, qkv, M, C3, CDIM);

  // 2. sampling coordinates
  coords_kernel<<<dim3(WN * WN, BATCH), 256, 0, stream>>>(
      x, conv_w, conv_b, coords);

  // 3. fused grid-sample + window attention (bf16 in/out)
  attn_kernel<<<dim3(NHEAD, WN * WN, BATCH), 256, 0, stream>>>(
      qkv, coords, attn_out);

  // 4. out = attn_out @ proj_w^T + proj_b   (f32 out)
  gemm_bf16_bias<float><<<dim3(CDIM / 32, M / 128), 256, 0, stream>>>(
      attn_out, pwb, proj_b, out, M, CDIM, CDIM);
}
